// PatchEmbedding_13769665151180
// MI455X (gfx1250) — matrix operand order repaired and checked by reading, not compiled, not run on hardware
//
#include <hip/hip_runtime.h>
#include <hip/hip_bf16.h>

// ViT patch embedding for MI455X (gfx1250, wave32, WMMA).
//   proj = x @ W + b  via v_wmma_f32_16x16x32_bf16 (bf16 in, fp32 accumulate)
//   out[b, 0]      = emb[0]   + pos[0]
//   out[b, 2i+1]   = emb[i+1] + pos[2i+1]
//   out[b, 2i+2]   = proj[b,i]+ pos[2i+2]
//
// d_ws holds W transposed+converted to bf16 ([E][K], 1.18 MB, L2-resident).
// Wave tile 32x64 = 2x4 WMMA tiles -> 8 WMMAs per 16 b128 loads.

#define BATCHN   128
#define PATCHES  196
#define KDIM     768
#define EDIM     768
#define MROWS    (BATCHN * PATCHES)   // 25088
#define TOKENS   (2 * PATCHES + 1)    // 393

typedef __attribute__((ext_vector_type(4)))  float  f32x4;
typedef __attribute__((ext_vector_type(8)))  float  f32x8;
typedef __attribute__((ext_vector_type(8)))  __bf16 bf16x8;
typedef __attribute__((ext_vector_type(16))) __bf16 bf16x16;

// ---------------------------------------------------------------------------
// Kernel 1: W [K][N] fp32 (row-major over N)  ->  Wt [N][K] bf16 (K-contiguous)
// Classic 32x32 LDS transpose: coalesced global reads and writes.
// ---------------------------------------------------------------------------
__global__ __launch_bounds__(256) void transpose_convert_w(
    const float* __restrict__ W, __bf16* __restrict__ Wt) {
  __shared__ float tile[32][33];
  const int n0 = blockIdx.x * 32;
  const int k0 = blockIdx.y * 32;
  const int tx = threadIdx.x;  // 0..31
  const int ty = threadIdx.y;  // 0..7
#pragma unroll
  for (int r = 0; r < 4; ++r) {
    const int k = k0 + ty + r * 8;
    tile[ty + r * 8][tx] = W[(size_t)k * EDIM + n0 + tx];
  }
  __syncthreads();
#pragma unroll
  for (int r = 0; r < 4; ++r) {
    const int n = n0 + ty + r * 8;
    Wt[(size_t)n * KDIM + k0 + tx] = (__bf16)tile[tx][ty + r * 8];
  }
}

// ---------------------------------------------------------------------------
// Kernel 2: GEMM + bias + pos, scattered into interleaved token slots.
// Block = 256 threads = 8 waves laid out 2(M) x 4(N); block tile 64x256.
// Wave tile 32x64 = 2x4 WMMA 16x16 tiles; K loop in steps of 32.
// ---------------------------------------------------------------------------
__global__ __launch_bounds__(256) void patch_proj_wmma(
    const float* __restrict__ x, const __bf16* __restrict__ Wt,
    const float* __restrict__ bias, const float* __restrict__ pos,
    float* __restrict__ out) {
  const int lane = threadIdx.x & 31;
  const int wave = threadIdx.x >> 5;
  const int wm = wave & 1;   // 0..1
  const int wn = wave >> 1;  // 0..3
  const int tileM = blockIdx.x * 64 + wm * 32;
  const int tileN = blockIdx.y * 256 + wn * 64;
  const int lmod = lane & 15;
  const bool hiHalf = lane >= 16;

  f32x8 acc[2][4] = {};

  // A fragment rows / B fragment columns owned by this lane.
  const int arow[2] = {tileM + lmod, tileM + 16 + lmod};
  int ncol[4];
  float bias_n[4];
#pragma unroll
  for (int j = 0; j < 4; ++j) {
    ncol[j] = tileN + j * 16 + lmod;
    bias_n[j] = bias[ncol[j]];
  }

  // ISA 7.12.2 fragment K-bases per lane half:
  //   A (16x32 bf16): lanes 0-15 hold K {0..7,16..23}; lanes 16-31 hold +8.
  //   B (32x16 bf16): lanes 0-15 hold K {0..7,16..23}; lanes 16-31 hold +8.
  const int kA = hiHalf ? 8 : 0;
  const int kB = hiHalf ? 8 : 0;

  for (int k0 = 0; k0 < KDIM; k0 += 32) {
    if (k0 + 32 < KDIM)  // global_prefetch_b8 of next A slab
      __builtin_prefetch(x + (size_t)arow[0] * KDIM + k0 + 32 + kA, 0, 1);

    bf16x16 af[2];
#pragma unroll
    for (int i = 0; i < 2; ++i) {
      const float* xp = x + (size_t)arow[i] * KDIM + k0 + kA;
      const f32x4 r0 = *(const f32x4*)(xp + 0);
      const f32x4 r1 = *(const f32x4*)(xp + 4);
      const f32x4 r2 = *(const f32x4*)(xp + 16);
      const f32x4 r3 = *(const f32x4*)(xp + 20);
      bf16x16 a;
#pragma unroll
      for (int e = 0; e < 4; ++e) {
        a[e + 0]  = (__bf16)r0[e];
        a[e + 4]  = (__bf16)r1[e];
        a[e + 8]  = (__bf16)r2[e];
        a[e + 12] = (__bf16)r3[e];
      }
      af[i] = a;
    }

    bf16x16 bfr[4];
#pragma unroll
    for (int j = 0; j < 4; ++j) {
      const __bf16* wp = Wt + (size_t)ncol[j] * KDIM + k0 + kB;
      const bf16x8 lo = *(const bf16x8*)(wp + 0);
      const bf16x8 hi = *(const bf16x8*)(wp + 16);
      bfr[j] = __builtin_shufflevector(lo, hi, 0, 1, 2, 3, 4, 5, 6, 7, 8, 9,
                                       10, 11, 12, 13, 14, 15);
    }

#pragma unroll
    for (int i = 0; i < 2; ++i)
#pragma unroll
      for (int j = 0; j < 4; ++j)
        acc[i][j] = __builtin_amdgcn_wmma_f32_16x16x32_bf16(
            false, af[i], false, bfr[j], (short)0, acc[i][j], false, false);
  }

  // Epilogue: C layout (ISA 7.12.2): VGPR v holds row v (lanes 0-15) or
  // row v+8 (lanes 16-31); column = lane%16. Fuse +bias +pos and scatter
  // to token 2p+2 with non-temporal stores (output is write-once).
#pragma unroll
  for (int i = 0; i < 2; ++i) {
    const int mbase = tileM + i * 16 + (hiHalf ? 8 : 0);
#pragma unroll
    for (int v = 0; v < 8; ++v) {
      const int m = mbase + v;
      const int bidx = m / PATCHES;
      const int p = m - bidx * PATCHES;
      const int tok = 2 * p + 2;
#pragma unroll
      for (int j = 0; j < 4; ++j) {
        const int n = ncol[j];
        const float val =
            acc[i][j][v] + bias_n[j] + pos[(size_t)tok * EDIM + n];
        __builtin_nontemporal_store(
            val, out + ((size_t)bidx * TOKENS + tok) * EDIM + n);
      }
    }
  }
}

// ---------------------------------------------------------------------------
// Kernel 3: broadcast cls + patch-embedding tokens (+pos). Pure streaming
// store (77 MB); sources are tiny and L2-resident. float4 NT stores.
// ---------------------------------------------------------------------------
__global__ __launch_bounds__(256) void broadcast_tokens(
    const float* __restrict__ emb, const float* __restrict__ pos,
    float* __restrict__ out) {
  const int total = BATCHN * (PATCHES + 1) * (EDIM / 4);
  const int tid = blockIdx.x * blockDim.x + threadIdx.x;
  if (tid >= total) return;
  const int n4 = tid % (EDIM / 4);
  const int rest = tid / (EDIM / 4);
  const int slot = rest % (PATCHES + 1);  // 0 = cls, 1..P = patches
  const int bidx = rest / (PATCHES + 1);
  const int tok = (slot == 0) ? 0 : (2 * slot - 1);
  const f32x4 e = *(const f32x4*)(emb + (size_t)slot * EDIM + n4 * 4);
  const f32x4 pz = *(const f32x4*)(pos + (size_t)tok * EDIM + n4 * 4);
  const f32x4 v = e + pz;
  __builtin_nontemporal_store(
      v, (f32x4*)(out + ((size_t)bidx * TOKENS + tok) * EDIM + n4 * 4));
}

extern "C" void kernel_launch(void* const* d_in, const int* in_sizes, int n_in,
                              void* d_out, int out_size, void* d_ws,
                              size_t ws_size, hipStream_t stream) {
  const float* x   = (const float*)d_in[0];  // [128,196,768]
  const float* W   = (const float*)d_in[1];  // [768,768]
  const float* b   = (const float*)d_in[2];  // [768]
  const float* emb = (const float*)d_in[3];  // [197,768]
  const float* pos = (const float*)d_in[4];  // [1,393,768]
  float* out = (float*)d_out;                // [128,393,768]
  __bf16* Wt = (__bf16*)d_ws;                // [768][768] bf16 = 1.18 MB

  // 1) one-time W transpose + bf16 convert into workspace
  transpose_convert_w<<<dim3(EDIM / 32, KDIM / 32), dim3(32, 8), 0, stream>>>(
      W, Wt);

  // 2) WMMA GEMM fused with bias/pos and interleaved scatter
  patch_proj_wmma<<<dim3(MROWS / 64, EDIM / 256), 256, 0, stream>>>(
      x, Wt, b, pos, out);

  // 3) broadcast embedding tokens (independent of GEMM output slots)
  const int total = BATCHN * (PATCHES + 1) * (EDIM / 4);
  broadcast_tokens<<<(total + 255) / 256, 256, 0, stream>>>(emb, pos, out);
}
